// FPTUNet_64665027608657
// MI455X (gfx1250) — matrix operand order repaired and checked by reading, not compiled, not run on hardware
//
#include <hip/hip_runtime.h>
#include <hip/hip_fp16.h>

typedef __attribute__((ext_vector_type(16))) _Float16 v16h;
typedef __attribute__((ext_vector_type(8)))  float    v8f;

#define NPTS   100000
#define MPAIR  1000000
#define CDIM   128
#define KVOL   27
#define NHEAD  8
#define EPS_BN   1e-5f
#define EPS_NORM 1e-12f

#if __has_builtin(__builtin_amdgcn_global_load_async_to_lds_b128)
#define HAVE_ASYNC_LDS 1
#else
#define HAVE_ASYNC_LDS 0
#endif

// ---------------- workspace layout (bytes) ----------------
constexpr size_t wal(size_t x) { return (x + 255u) & ~(size_t)255u; }
constexpr size_t OFF_T1     = 0;                                           // N*3 f32
constexpr size_t OFF_STATS1 = wal(OFF_T1 + (size_t)NPTS * 3 * 4);          // 16 f32
constexpr size_t OFF_STATS2 = wal(OFF_STATS1 + 16 * 4);                    // 512 f32
constexpr size_t OFF_WF     = wal(OFF_STATS2 + 512 * 4);                   // 4*8192 dwords (f16 frags)
constexpr size_t OFF_PN     = wal(OFF_WF + 4 * 8192 * 4);                  // 27*128 f32
constexpr size_t OFF_T2     = wal(OFF_PN + (size_t)KVOL * CDIM * 4);       // N*128 f32 (aliased as ACC)
constexpr size_t OFF_QN     = wal(OFF_T2 + (size_t)NPTS * CDIM * 4);       // N*128 f16
constexpr size_t OFF_V      = wal(OFF_QN + (size_t)NPTS * CDIM * 2);       // N*128 f16
constexpr size_t OFF_ACC    = OFF_T2;                                      // alias (t2 consumed first)

// ---------------- helpers ----------------
__device__ inline unsigned short f2h_bits(float x) {
    union { _Float16 h; unsigned short s; } u; u.h = (_Float16)x; return u.s;
}

// Pointer types matching the async-to-LDS builtin's prototype:
// param0: AS(1) int-vec4*  (global source), param1: AS(3) int-vec4* (LDS dest)
typedef int b128i __attribute__((vector_size(16)));
typedef __attribute__((address_space(1))) b128i* g_b128p;
typedef __attribute__((address_space(3))) b128i* l_b128p;

// Async DMA of a 16x128 f32 tile (8KB) global -> LDS. Per lane 16 x b128 copies,
// tracked by ASYNCcnt; falls back to a synchronous ds-store copy if the builtin
// is not available on this toolchain.
__device__ inline void tile_copy_f32(const float* __restrict__ g, float* l, int lane) {
#if HAVE_ASYNC_LDS
#pragma unroll
    for (int it = 0; it < 16; ++it) {
        int idx = (it * 32 + lane) * 4;
        unsigned long long ga = (unsigned long long)(const void*)(g + idx);
        unsigned           la = (unsigned)(unsigned long long)(const void*)(l + idx); // low 32b = LDS offset
        __builtin_amdgcn_global_load_async_to_lds_b128((g_b128p)ga, (l_b128p)(unsigned long long)la, 0, 0);
    }
#else
#pragma unroll
    for (int it = 0; it < 16; ++it) {
        int idx = (it * 32 + lane) * 4;
        *(float4*)(l + idx) = *(const float4*)(g + idx);
    }
#endif
}

__device__ inline void tile_copy_wait() {
#if HAVE_ASYNC_LDS
#if __has_builtin(__builtin_amdgcn_s_wait_asynccnt)
    __builtin_amdgcn_s_wait_asynccnt(0);
#else
    asm volatile("s_wait_asynccnt 0x0" ::: "memory");
#endif
#endif
}

// A fragment (16x32 f16, M x K) from an LDS tile stored row-major [16][128] halves.
__device__ inline v16h load_a_frag(const _Float16* lds, int kt, int lane) {
    union { v16h h; unsigned u[8]; } fr;
    const unsigned* l32 = (const unsigned*)lds;
    int m  = lane & 15;
    int hi = (lane >> 4) & 1;
    int rowbase = m * CDIM;
#pragma unroll
    for (int p = 0; p < 8; ++p) {
        int k = kt * 32 + ((p >= 4) ? 16 : 0) + hi * 8 + 2 * (p & 3);
        fr.u[p] = l32[(rowbase + k) >> 1];
    }
    return fr.h;
}

// B fragment (32x16 f16, K x N) from pre-swizzled weight buffer.
__device__ inline v16h load_b_frag(const unsigned* wf, int kt, int nt, int lane) {
    union { v16h h; uint4 q[2]; } fr;
    const uint4* p = (const uint4*)(wf + (size_t)(((kt * 8 + nt) * 32 + lane)) * 8);
    fr.q[0] = p[0];
    fr.q[1] = p[1];
    return fr.h;
}

// ---------------- utility kernels ----------------
__global__ void k_zero(float* p, long n) {
    long i = (long)blockIdx.x * blockDim.x + threadIdx.x;
    long stride = (long)gridDim.x * blockDim.x;
    for (; i < n; i += stride) p[i] = 0.0f;
}

// Pre-swizzle w3/wq/wv/wo (f32 row-major [128][128]) into f16 B-fragment layout.
__global__ void k_prep_w(const float* __restrict__ w3, const float* __restrict__ wq,
                         const float* __restrict__ wv, const float* __restrict__ wo,
                         unsigned* __restrict__ wf) {
    int idx = blockIdx.x * blockDim.x + threadIdx.x;      // 4*8192 dwords
    if (idx >= 4 * 8192) return;
    int wsel = idx >> 13;
    int r    = idx & 8191;
    int d    = r & 7;
    int lane = (r >> 3) & 31;
    int nt   = (r >> 8) & 7;
    int kt   = (r >> 11) & 3;
    const float* W = (wsel == 0) ? w3 : (wsel == 1) ? wq : (wsel == 2) ? wv : wo;
    int n = nt * 16 + (lane & 15);
    int k = kt * 32 + ((d >= 4) ? 16 : 0) + ((lane >> 4) & 1) * 8 + 2 * (d & 3);
    unsigned lo = f2h_bits(W[(size_t)k * CDIM + n]);
    unsigned hi = f2h_bits(W[(size_t)(k + 1) * CDIM + n]);
    wf[idx] = lo | (hi << 16);
}

// Normalize pos_enc [27][8][16] -> pn [27][128] f32.
__global__ void k_pos(const float* __restrict__ pe, float* __restrict__ pn) {
    int t = blockIdx.x * blockDim.x + threadIdx.x;
    if (t >= KVOL * NHEAD) return;
    const float* src = pe + (size_t)t * 16;
    float v[16], ss = 0.f;
#pragma unroll
    for (int i = 0; i < 16; ++i) { v[i] = src[i]; ss += v[i] * v[i]; }
    float inv = 1.0f / fmaxf(sqrtf(ss), EPS_NORM);
    float* dst = pn + (size_t)(t >> 3) * CDIM + (t & 7) * 16;
#pragma unroll
    for (int i = 0; i < 16; ++i) dst[i] = v[i] * inv;
}

// t1 = points @ w1 (N x 3) + channel sum/sumsq for BN1.
__global__ __launch_bounds__(256) void k_stage1(const float* __restrict__ pts,
                                                const float* __restrict__ w1,
                                                float* __restrict__ t1, float* __restrict__ stats1) {
    __shared__ float ssum[3], ssq[3];
    if (threadIdx.x < 3) { ssum[threadIdx.x] = 0.f; ssq[threadIdx.x] = 0.f; }
    __syncthreads();
    int i = blockIdx.x * blockDim.x + threadIdx.x;
    if (i < NPTS) {
        float p0 = pts[i * 3 + 0], p1 = pts[i * 3 + 1], p2 = pts[i * 3 + 2];
#pragma unroll
        for (int j = 0; j < 3; ++j) {
            float o = p0 * w1[j] + p1 * w1[3 + j] + p2 * w1[6 + j];
            t1[i * 3 + j] = o;
            atomicAdd(&ssum[j], o);
            atomicAdd(&ssq[j], o * o);
        }
    }
    __syncthreads();
    if (threadIdx.x < 3) {
        atomicAdd(&stats1[threadIdx.x], ssum[threadIdx.x]);
        atomicAdd(&stats1[3 + threadIdx.x], ssq[threadIdx.x]);
    }
}

__global__ void k_bn1(const float* __restrict__ g1, const float* __restrict__ b1,
                      float* __restrict__ stats1) {
    int j = threadIdx.x;
    if (j >= 3) return;
    float mean = stats1[j] / (float)NPTS;
    float var  = stats1[3 + j] / (float)NPTS - mean * mean;
    float sc   = g1[j] * rsqrtf(var + EPS_BN);
    stats1[6 + j] = sc;
    stats1[9 + j] = b1[j] - mean * sc;
}

// h1 = relu(bn1(t1)); t2 = h1 @ w2 (N x 128); accumulate BN2 stats. One wave per point.
__global__ __launch_bounds__(256) void k_stage2(const float* __restrict__ t1,
                                                const float* __restrict__ stats1,
                                                const float* __restrict__ w2,
                                                float* __restrict__ t2, float* __restrict__ stats2) {
    int lane = threadIdx.x & 31;
    int wid  = blockIdx.x * (blockDim.x >> 5) + (threadIdx.x >> 5);
    int nw   = gridDim.x * (blockDim.x >> 5);
    float s0 = stats1[6], s1 = stats1[7], s2 = stats1[8];
    float d0 = stats1[9], d1 = stats1[10], d2 = stats1[11];
    float sum[4] = {0, 0, 0, 0}, sq[4] = {0, 0, 0, 0};
    for (int i = wid; i < NPTS; i += nw) {
        float a0 = fmaxf(t1[i * 3 + 0] * s0 + d0, 0.f);
        float a1 = fmaxf(t1[i * 3 + 1] * s1 + d1, 0.f);
        float a2 = fmaxf(t1[i * 3 + 2] * s2 + d2, 0.f);
#pragma unroll
        for (int r = 0; r < 4; ++r) {
            int c = lane + r * 32;
            float v = a0 * w2[c] + a1 * w2[CDIM + c] + a2 * w2[2 * CDIM + c];
            t2[(size_t)i * CDIM + c] = v;
            sum[r] += v; sq[r] += v * v;
        }
    }
#pragma unroll
    for (int r = 0; r < 4; ++r) {
        int c = lane + r * 32;
        atomicAdd(&stats2[c], sum[r]);
        atomicAdd(&stats2[CDIM + c], sq[r]);
    }
}

__global__ void k_bn2(const float* __restrict__ g2, const float* __restrict__ b2,
                      float* __restrict__ stats2) {
    int c = threadIdx.x;
    if (c >= CDIM) return;
    float mean = stats2[c] / (float)NPTS;
    float var  = stats2[CDIM + c] / (float)NPTS - mean * mean;
    float sc   = g2[c] * rsqrtf(var + EPS_BN);
    stats2[256 + c] = sc;
    stats2[384 + c] = b2[c] - mean * sc;
}

// Fused: h2=relu(bn2(t2)); X=feats + h2@W3 + b3; Q=X@Wq+bq (l2norm per head); V=X@Wv+bv.
// One wave per 16-row tile; feats tile staged via async DMA, 3 WMMA GEMMs per wave.
__global__ __launch_bounds__(128) void k_fuse_qv(
    const float* __restrict__ t2, const float* __restrict__ feats,
    const float* __restrict__ stats2, const float* __restrict__ b3,
    const float* __restrict__ bq, const float* __restrict__ bv,
    const unsigned* __restrict__ w3f, const unsigned* __restrict__ wqf,
    const unsigned* __restrict__ wvf,
    unsigned short* __restrict__ qn_out, unsigned short* __restrict__ v_out) {
    __shared__ _Float16 ldsH[4][16 * CDIM];
    __shared__ _Float16 ldsX[4][16 * CDIM];
    __shared__ float    ldsF[4][16 * CDIM];
    int wave = threadIdx.x >> 5;
    int lane = threadIdx.x & 31;
    int tile = blockIdx.x * 4 + wave;
    if (tile >= NPTS / 16) return;                  // wave-uniform; no barriers used
    int row0 = tile * 16;
    int hi   = (lane >> 4) & 1;
    int ncol = lane & 15;
    const float* scale2 = stats2 + 256;
    const float* shift2 = stats2 + 384;

    // Kick off async feats tile DMA first so it overlaps BN staging and A-fragment loads.
    float* lf = ldsF[wave];
    tile_copy_f32(feats + (size_t)row0 * CDIM, lf, lane);

    _Float16* lh = ldsH[wave];
    for (int idx = lane; idx < 16 * CDIM; idx += 32) {
        int m = idx >> 7, c = idx & 127;
        float v = t2[(size_t)(row0 + m) * CDIM + c];
        v = fmaxf(v * scale2[c] + shift2[c], 0.f);
        lh[idx] = (_Float16)v;
    }
    v16h ah[4];
#pragma unroll
    for (int kt = 0; kt < 4; ++kt) ah[kt] = load_a_frag(lh, kt, lane);

    tile_copy_wait();                               // feats tile resident in LDS

    // GEMM1: X = feats + h2@W3 + b3 -> ldsX (f16)
    _Float16* lx = ldsX[wave];
#pragma unroll
    for (int nt = 0; nt < 8; ++nt) {
        v8f acc = {};
#pragma unroll
        for (int kt = 0; kt < 4; ++kt) {
            v16h b = load_b_frag(w3f, kt, nt, lane);
            acc = __builtin_amdgcn_wmma_f32_16x16x32_f16(false, ah[kt], false, b,
                                                         (short)0, acc, false, false);
        }
        int n = nt * 16 + ncol;
        float bias = b3[n];
        float fv[8];
#pragma unroll
        for (int r = 0; r < 8; ++r) fv[r] = lf[(r + hi * 8) * CDIM + n];
#pragma unroll
        for (int r = 0; r < 8; ++r) {
            int m = r + hi * 8;
            lx[m * CDIM + n] = (_Float16)(acc[r] + bias + fv[r]);
        }
    }
    v16h ax[4];
#pragma unroll
    for (int kt = 0; kt < 4; ++kt) ax[kt] = load_a_frag(lx, kt, lane);

    // GEMM2: Q (nt == head; l2-normalize across the 16 lanes holding one row)
#pragma unroll
    for (int nt = 0; nt < 8; ++nt) {
        v8f acc = {};
#pragma unroll
        for (int kt = 0; kt < 4; ++kt) {
            v16h b = load_b_frag(wqf, kt, nt, lane);
            acc = __builtin_amdgcn_wmma_f32_16x16x32_f16(false, ax[kt], false, b,
                                                         (short)0, acc, false, false);
        }
        int n = nt * 16 + ncol;
        float bias = bq[n];
#pragma unroll
        for (int r = 0; r < 8; ++r) {
            float q = acc[r] + bias;
            float ss = q * q;
            ss += __shfl_xor(ss, 1, 32);
            ss += __shfl_xor(ss, 2, 32);
            ss += __shfl_xor(ss, 4, 32);
            ss += __shfl_xor(ss, 8, 32);
            float inv = 1.0f / fmaxf(sqrtf(ss), EPS_NORM);
            int m = r + hi * 8;
            qn_out[(size_t)(row0 + m) * CDIM + n] = f2h_bits(q * inv);
        }
    }
    // GEMM3: V
#pragma unroll
    for (int nt = 0; nt < 8; ++nt) {
        v8f acc = {};
#pragma unroll
        for (int kt = 0; kt < 4; ++kt) {
            v16h b = load_b_frag(wvf, kt, nt, lane);
            acc = __builtin_amdgcn_wmma_f32_16x16x32_f16(false, ax[kt], false, b,
                                                         (short)0, acc, false, false);
        }
        int n = nt * 16 + ncol;
        float bias = bv[n];
#pragma unroll
        for (int r = 0; r < 8; ++r) {
            int m = r + hi * 8;
            v_out[(size_t)(row0 + m) * CDIM + n] = f2h_bits(acc[r] + bias);
        }
    }
}

// attn[m,h] = <qn[q_idx],pn[kidx]>; acc[q_idx] += attn*v[k_idx]. One wave per pair;
// lane owns 4 channels; contiguous per-wave pair ranges keep index loads line-local.
__global__ __launch_bounds__(256) void k_attn(
    const int* __restrict__ q_idx, const int* __restrict__ k_idx,
    const int* __restrict__ kern_idx,
    const unsigned short* __restrict__ qn, const unsigned short* __restrict__ vv,
    const float* __restrict__ pn, float* __restrict__ acc) {
    int lane = threadIdx.x & 31;
    int wid  = blockIdx.x * (blockDim.x >> 5) + (threadIdx.x >> 5);
    int nw   = gridDim.x * (blockDim.x >> 5);
    int per  = (MPAIR + nw - 1) / nw;
    int m0   = wid * per;
    int m1   = (m0 + per < MPAIR) ? (m0 + per) : MPAIR;
    int c0   = lane * 4;
    for (int m = m0; m < m1; ++m) {
        int qi = q_idx[m], ki = k_idx[m], kk = kern_idx[m];
        const __half2* qp = (const __half2*)(qn + (size_t)qi * CDIM + c0);
        float2 q01 = __half22float2(qp[0]);
        float2 q23 = __half22float2(qp[1]);
        float4 p4  = *(const float4*)(pn + (size_t)kk * CDIM + c0);
        float part = q01.x * p4.x + q01.y * p4.y + q23.x * p4.z + q23.y * p4.w;
        part += __shfl_xor(part, 1, 32);
        part += __shfl_xor(part, 2, 32);   // head dot in every lane of the 4-lane group
        const __half2* vp = (const __half2*)(vv + (size_t)ki * CDIM + c0);
        float2 v01 = __half22float2(vp[0]);
        float2 v23 = __half22float2(vp[1]);
        float* ap = acc + (size_t)qi * CDIM + c0;
        atomicAdd(ap + 0, part * v01.x);
        atomicAdd(ap + 1, part * v01.y);
        atomicAdd(ap + 2, part * v23.x);
        atomicAdd(ap + 3, part * v23.y);
    }
}

// final = acc @ wo + bo  (WMMA, one wave per 16-row tile)
__global__ __launch_bounds__(128) void k_out(const float* __restrict__ acc_in,
                                             const unsigned* __restrict__ wof,
                                             const float* __restrict__ bo,
                                             float* __restrict__ out) {
    __shared__ _Float16 ldsA[4][16 * CDIM];
    int wave = threadIdx.x >> 5;
    int lane = threadIdx.x & 31;
    int tile = blockIdx.x * 4 + wave;
    if (tile >= NPTS / 16) return;
    int row0 = tile * 16;
    int hi   = (lane >> 4) & 1;
    int ncol = lane & 15;
    _Float16* la = ldsA[wave];
    for (int idx = lane; idx < 16 * CDIM; idx += 32) {
        int m = idx >> 7, c = idx & 127;
        la[idx] = (_Float16)acc_in[(size_t)(row0 + m) * CDIM + c];
    }
    v16h af[4];
#pragma unroll
    for (int kt = 0; kt < 4; ++kt) af[kt] = load_a_frag(la, kt, lane);
#pragma unroll
    for (int nt = 0; nt < 8; ++nt) {
        v8f acc = {};
#pragma unroll
        for (int kt = 0; kt < 4; ++kt) {
            v16h b = load_b_frag(wof, kt, nt, lane);
            acc = __builtin_amdgcn_wmma_f32_16x16x32_f16(false, af[kt], false, b,
                                                         (short)0, acc, false, false);
        }
        int n = nt * 16 + ncol;
        float bias = bo[n];
#pragma unroll
        for (int r = 0; r < 8; ++r) {
            int m = r + hi * 8;
            out[(size_t)(row0 + m) * CDIM + n] = acc[r] + bias;
        }
    }
}

extern "C" void kernel_launch(void* const* d_in, const int* in_sizes, int n_in,
                              void* d_out, int out_size, void* d_ws, size_t ws_size,
                              hipStream_t stream) {
    const float* points   = (const float*)d_in[0];
    const float* feats    = (const float*)d_in[1];
    const int*   k_idx    = (const int*)d_in[2];
    const int*   q_idx    = (const int*)d_in[3];
    const int*   kern_idx = (const int*)d_in[4];
    const float* w1 = (const float*)d_in[5];
    const float* g1 = (const float*)d_in[6];
    const float* b1 = (const float*)d_in[7];
    const float* w2 = (const float*)d_in[8];
    const float* g2 = (const float*)d_in[9];
    const float* b2 = (const float*)d_in[10];
    const float* w3 = (const float*)d_in[11];
    const float* b3 = (const float*)d_in[12];
    const float* wq = (const float*)d_in[13];
    const float* bq = (const float*)d_in[14];
    const float* wv = (const float*)d_in[15];
    const float* bv = (const float*)d_in[16];
    const float* wo = (const float*)d_in[17];
    const float* bo = (const float*)d_in[18];
    const float* pe = (const float*)d_in[19];

    char* wsb = (char*)d_ws;
    float*          t1     = (float*)(wsb + OFF_T1);
    float*          stats1 = (float*)(wsb + OFF_STATS1);
    float*          stats2 = (float*)(wsb + OFF_STATS2);
    unsigned*       wf     = (unsigned*)(wsb + OFF_WF);
    float*          pn     = (float*)(wsb + OFF_PN);
    float*          t2     = (float*)(wsb + OFF_T2);
    unsigned short* qn     = (unsigned short*)(wsb + OFF_QN);
    unsigned short* vbuf   = (unsigned short*)(wsb + OFF_V);
    float*          accum  = (float*)(wsb + OFF_ACC);   // aliases t2 (consumed before zeroing)

    const int TILE_BLOCKS = (NPTS / 16 + 3) / 4;        // 1563 blocks, 4 waves each

    k_zero<<<1, 256, 0, stream>>>(stats1, 16);
    k_zero<<<2, 256, 0, stream>>>(stats2, 512);
    k_prep_w<<<128, 256, 0, stream>>>(w3, wq, wv, wo, wf);
    k_pos<<<1, 256, 0, stream>>>(pe, pn);
    k_stage1<<<(NPTS + 255) / 256, 256, 0, stream>>>(points, w1, t1, stats1);
    k_bn1<<<1, 32, 0, stream>>>(g1, b1, stats1);
    k_stage2<<<256, 256, 0, stream>>>(t1, stats1, w2, t2, stats2);
    k_bn2<<<1, 128, 0, stream>>>(g2, b2, stats2);
    k_fuse_qv<<<TILE_BLOCKS, 128, 0, stream>>>(t2, feats, stats2, b3, bq, bv,
                                               wf, wf + 8192, wf + 16384, qn, vbuf);
    k_zero<<<2048, 256, 0, stream>>>(accum, (long)NPTS * CDIM);
    k_attn<<<1024, 256, 0, stream>>>(q_idx, k_idx, kern_idx, qn, vbuf, pn, accum);
    k_out<<<TILE_BLOCKS, 128, 0, stream>>>(accum, wf + 24576, bo, (float*)d_out);
}
